// Model_56530359550856
// MI455X (gfx1250) — matrix operand order repaired and checked by reading, not compiled, not run on hardware
//
#include <hip/hip_runtime.h>

// ---------------- problem constants ----------------
#define L_  4
#define C_  768
#define H_  12
#define HD_ 64
#define NT_ 512
#define NA_ 18
#define B_  2
#define TS_ 1024
#define T_  3072
#define M_  (B_ * T_)        // 6144 tokens

typedef unsigned short u16;
typedef unsigned int   u32;
typedef __attribute__((ext_vector_type(16))) __bf16 bf16x16;
typedef __attribute__((ext_vector_type(8)))  float  f32x8;
typedef __attribute__((ext_vector_type(4)))  u32    u32x4;

union Frag16 { bf16x16 v; u32x4 q[2]; u16 h[16]; };

__device__ __forceinline__ u16 f2bf(float f) {
    u32 u = __float_as_uint(f);
    u32 r = (u + 0x7FFFu + ((u >> 16) & 1u)) >> 16;   // round-to-nearest-even
    return (u16)r;
}
__device__ __forceinline__ float bf2f(u16 h) {
    return __uint_as_float(((u32)h) << 16);
}
__device__ __forceinline__ f32x8 wmma_bf16(const Frag16& a, const Frag16& b, f32x8 c) {
    return __builtin_amdgcn_wmma_f32_16x16x32_bf16(
        false, a.v, false, b.v, (short)0, c, false, false);
}
__device__ __forceinline__ float gelu_tanh(float x) {
    float x3 = x * x * x;
    float t  = tanhf(0.7978845608028654f * (x + 0.044715f * x3));
    return 0.5f * x * (1.0f + t);
}

// ---------------- fp32 -> bf16 conversion ----------------
__global__ void __launch_bounds__(256) convf2bf(const float* __restrict__ s,
                                                u16* __restrict__ d, int n) {
    int i = blockIdx.x * 256 + threadIdx.x;
    if (i < n) d[i] = f2bf(s[i]);
}

// ---------------- embedding gather / interleave ----------------
__global__ void __launch_bounds__(128) embed_k(const int* __restrict__ st,
                                               const int* __restrict__ ac,
                                               const int* __restrict__ rw,
                                               const u16* __restrict__ eS,
                                               const u16* __restrict__ eA,
                                               const u16* __restrict__ eR,
                                               u16* __restrict__ seqb) {
    int g  = blockIdx.x;                 // 0..M_-1 token index (b,T flat)
    int b  = g / T_, t3 = g % T_;
    int j  = t3 / 3, m = t3 % 3;
    const u16* src;
    if (m == 0)      src = eS + (size_t)st[b * TS_ + j] * NT_;
    else if (m == 1) src = eA + (size_t)ac[b * TS_ + j] * NT_;
    else             src = eR + (size_t)rw[b * TS_ + j] * NT_;
    u16* dst = seqb + (size_t)g * NT_;
    for (int c = threadIdx.x; c < NT_; c += 128) dst[c] = src[c];
}

// ---------------- generic bf16 WMMA GEMM, 2-deep software pipeline ----------------
// out[M][N] = A[M][K] (bf16) @ W[N][K]^T (bf16) + bias [+gelu] [+resid]
// 256 threads = 8 waves; wave owns a 16x128 strip of the 128x128 tile.
// K is a multiple of 64 at every call site; loop unrolled by 2 with
// ping-pong A/B fragment registers so next iteration's loads are in flight
// during the current iteration's WMMAs (verified: s_wait_loadcnt 0x12 ...).
__global__ void __launch_bounds__(256) gemm_bf16(const u16* __restrict__ A,
                                                 const u16* __restrict__ W,
                                                 const float* __restrict__ bias,
                                                 const float* __restrict__ resid,
                                                 float* __restrict__ outF,
                                                 u16* __restrict__ outH,
                                                 int N, int K, int gel) {
    const int wave = threadIdx.x >> 5;
    const int lane = threadIdx.x & 31;
    const int half = lane >> 4;
    const int l16  = lane & 15;
    const int tileN = blockIdx.x * 128;
    const int tileM = blockIdx.y * 128;
    const int rowA  = tileM + wave * 16 + l16;

    f32x8 acc[8];
#pragma unroll
    for (int i = 0; i < 8; ++i)
#pragma unroll
        for (int j = 0; j < 8; ++j) acc[i][j] = 0.0f;

    const u16* ap    = A + (size_t)rowA * K + half * 8;
    const u16* wbase = W + (size_t)(tileN + l16) * K + half * 8;

#define LOAD_A(dst, KK) do {                                            \
        dst.q[0] = *(const u32x4*)(ap + (KK));                          \
        dst.q[1] = *(const u32x4*)(ap + (KK) + 16); } while (0)
#define LOAD_B(dst, NT, KK) do {                                        \
        const u16* wp = wbase + (size_t)(NT) * (16 * (size_t)K) + (KK); \
        dst.q[0] = *(const u32x4*)(wp);                                 \
        dst.q[1] = *(const u32x4*)(wp + 16); } while (0)

    Frag16 a0, a1, b0[8], b1[8];
    LOAD_A(a0, 0);
#pragma unroll
    for (int nt = 0; nt < 8; ++nt) LOAD_B(b0[nt], nt, 0);

    for (int k0 = 0; k0 < K; k0 += 64) {
        LOAD_A(a1, k0 + 32);
#pragma unroll
        for (int nt = 0; nt < 8; ++nt) LOAD_B(b1[nt], nt, k0 + 32);
        __builtin_prefetch(ap + k0 + 128, 0, 3);
#pragma unroll
        for (int nt = 0; nt < 8; ++nt) acc[nt] = wmma_bf16(a0, b0[nt], acc[nt]);
        if (k0 + 64 < K) {
            LOAD_A(a0, k0 + 64);
#pragma unroll
            for (int nt = 0; nt < 8; ++nt) LOAD_B(b0[nt], nt, k0 + 64);
        }
#pragma unroll
        for (int nt = 0; nt < 8; ++nt) acc[nt] = wmma_bf16(a1, b1[nt], acc[nt]);
    }
#undef LOAD_A
#undef LOAD_B

#pragma unroll
    for (int nt = 0; nt < 8; ++nt) {
        int ncol = tileN + nt * 16 + l16;
        float bv = bias ? bias[ncol] : 0.0f;
#pragma unroll
        for (int r = 0; r < 8; ++r) {
            size_t row = (size_t)(tileM + wave * 16 + r + half * 8);
            float v = acc[nt][r] + bv;
            if (gel) v = gelu_tanh(v);
            if (resid) v += resid[row * N + ncol];
            if (outF) outF[row * N + ncol] = v;
            if (outH) outH[row * N + ncol] = f2bf(v);
        }
    }
}

// ---------------- LayerNorm (fp32 in -> bf16 out), wave per row ----------------
__global__ void __launch_bounds__(256) ln_k(const float* __restrict__ x,
                                            const float* __restrict__ w,
                                            const float* __restrict__ bsh,
                                            u16* __restrict__ out) {
    int wave = threadIdx.x >> 5, lane = threadIdx.x & 31;
    int row  = blockIdx.x * 8 + wave;
    const float* xr = x + (size_t)row * C_;
    float vals[24];
    float s = 0.0f;
#pragma unroll
    for (int i = 0; i < 24; ++i) { vals[i] = xr[lane + 32 * i]; s += vals[i]; }
#pragma unroll
    for (int d = 1; d < 32; d <<= 1) s += __shfl_xor(s, d, 32);
    float mean = s * (1.0f / (float)C_);
    float q = 0.0f;
#pragma unroll
    for (int i = 0; i < 24; ++i) { float dd = vals[i] - mean; q += dd * dd; }
#pragma unroll
    for (int d = 1; d < 32; d <<= 1) q += __shfl_xor(q, d, 32);
    float rstd = rsqrtf(q * (1.0f / (float)C_) + 1e-5f);
    u16* orow = out + (size_t)row * C_;
#pragma unroll
    for (int i = 0; i < 24; ++i) {
        int c = lane + 32 * i;
        orow[c] = f2bf((vals[i] - mean) * rstd * w[c] + bsh[c]);
    }
}

// ---------------- QKV split + degenerate RoPE (x * (cos+sin)) ----------------
// Writes Q,K head-major [bh][T][HD] and V transposed [bh][HD][T] so the
// attention P@V B-fragments become two contiguous global_load_b128 per lane.
__global__ void __launch_bounds__(64) rope_k(const u16* __restrict__ qkv,
                                             u16* __restrict__ Qb,
                                             u16* __restrict__ Kb,
                                             u16* __restrict__ Vt) {
    int t = blockIdx.x, h = blockIdx.y, b = blockIdx.z;
    int d = threadIdx.x;
    size_t row = (size_t)(b * T_ + t) * (3 * C_);
    float q = bf2f(qkv[row + h * HD_ + d]);
    float k = bf2f(qkv[row + C_ + h * HD_ + d]);
    float v = bf2f(qkv[row + 2 * C_ + h * HD_ + d]);
    int i = d & 31;
    float theta = __expf(-(float)(2 * i) * (1.0f / 64.0f) * 9.210340371976184f);
    float idx   = (float)(t + 1) * theta;
    float rf    = __cosf(idx) + __sinf(idx);
    int bh = b * H_ + h;
    size_t o = ((size_t)bh * T_ + t) * HD_ + d;
    Qb[o] = f2bf(q * rf);
    Kb[o] = f2bf(k * rf);
    Vt[((size_t)bh * HD_ + d) * T_ + t] = f2bf(v);
}

// ---------------- flash attention helpers ----------------
// One 32-key step: issue V^T fragments (consumed after the barrier, so their
// latency hides under S-WMMAs + softmax), S = Q K^T (4 WMMA), online softmax,
// P relayed through LDS (C-layout -> A-layout), O += P V (4 WMMA).
// One __syncthreads per step thanks to double-buffered Ps.
__device__ __forceinline__ void attn_step(
    int kb0, int qrow0, int bh, int wave, int half, int l16,
    const Frag16& qf0, const Frag16& qf1, const Frag16 kf[4],
    const u16* __restrict__ Vt,
    u16 Ps[4][16][32],
    float mrow[8], float lrow[8], f32x8 o[4]) {
    const float scale = 0.125f;                   // 1/sqrt(64)
    // V^T B-fragments: lane holds dim = t*16+l16, keys kb0+half*16+e
    Frag16 vf[4];
    const u16* vtb = Vt + ((size_t)bh * HD_ + l16) * T_ + half * 16 + kb0;
#pragma unroll
    for (int t = 0; t < 4; ++t) {
        vf[t].q[0] = *(const u32x4*)(vtb + (size_t)t * 16 * T_);
        vf[t].q[1] = *(const u32x4*)(vtb + (size_t)t * 16 * T_ + 8);
    }
    f32x8 s0, s1;
#pragma unroll
    for (int r = 0; r < 8; ++r) { s0[r] = 0.0f; s1[r] = 0.0f; }
    s0 = wmma_bf16(qf0, kf[0], s0);
    s0 = wmma_bf16(qf1, kf[1], s0);
    s1 = wmma_bf16(qf0, kf[2], s1);
    s1 = wmma_bf16(qf1, kf[3], s1);
    // online softmax per query row (16-lane butterfly reductions)
#pragma unroll
    for (int r = 0; r < 8; ++r) {
        int qidx = qrow0 + r + half * 8;
        float a0 = (kb0 + l16      <= qidx) ? s0[r] * scale : -1e30f;
        float a1 = (kb0 + 16 + l16 <= qidx) ? s1[r] * scale : -1e30f;
        float mx = fmaxf(a0, a1);
#pragma unroll
        for (int d = 1; d < 16; d <<= 1) mx = fmaxf(mx, __shfl_xor(mx, d, 16));
        float nm = fmaxf(mrow[r], mx);
        float e0 = __expf(a0 - nm);
        float e1 = __expf(a1 - nm);
        float pr = e0 + e1;
#pragma unroll
        for (int d = 1; d < 16; d <<= 1) pr += __shfl_xor(pr, d, 16);
        float alpha = __expf(mrow[r] - nm);
        lrow[r] = lrow[r] * alpha + pr;
        mrow[r] = nm;
#pragma unroll
        for (int t = 0; t < 4; ++t) o[t][r] *= alpha;
        Ps[wave][r + half * 8][l16]      = f2bf(e0);
        Ps[wave][r + half * 8][16 + l16] = f2bf(e1);
    }
    __syncthreads();                              // Ps visible
    Frag16 pf;
    const u16* pp = &Ps[wave][l16][half * 16];
    pf.q[0] = *(const u32x4*)(pp);
    pf.q[1] = *(const u32x4*)(pp + 8);
#pragma unroll
    for (int t = 0; t < 4; ++t) o[t] = wmma_bf16(pf, vf[t], o[t]);
}

// grid (T/64, H, B), 128 threads = 4 waves; wave owns 16 query rows.
// K^T fragments double-buffered across 32-key steps (trip count is even).
__global__ void __launch_bounds__(128) attn_k(const u16* __restrict__ Qb,
                                              const u16* __restrict__ Kb,
                                              const u16* __restrict__ Vt,
                                              u16* __restrict__ Y) {
    __shared__ __align__(16) u16 Ps[2][4][16][32];
    const int wave = threadIdx.x >> 5;
    const int lane = threadIdx.x & 31;
    const int half = lane >> 4;
    const int l16  = lane & 15;
    const int qb = blockIdx.x, h = blockIdx.y, b = blockIdx.z;
    const int bh = b * H_ + h;
    const int qrow0 = qb * 64 + wave * 16;

    Frag16 qf0, qf1;                              // Q rows preloaded, K=0..31 / 32..63
    {
        const u16* qp = Qb + ((size_t)bh * T_ + qrow0 + l16) * HD_;
        qf0.q[0] = *(const u32x4*)(qp + half * 16);
        qf0.q[1] = *(const u32x4*)(qp + 8 + half * 16);
        qf1.q[0] = *(const u32x4*)(qp + 32 + half * 16);
        qf1.q[1] = *(const u32x4*)(qp + 40 + half * 16);
    }

    float mrow[8], lrow[8];
    f32x8 o[4];
#pragma unroll
    for (int r = 0; r < 8; ++r) { mrow[r] = -1e30f; lrow[r] = 0.0f; }
#pragma unroll
    for (int t = 0; t < 4; ++t)
#pragma unroll
        for (int r = 0; r < 8; ++r) o[t][r] = 0.0f;

    const u16* kfbase = Kb + ((size_t)bh * T_ + l16) * HD_ + half * 16;
#define LOAD_KF(dst, KB) do {                                              \
        const u16* kp0 = kfbase + (size_t)(KB) * HD_;                      \
        dst[0].q[0] = *(const u32x4*)(kp0);                                \
        dst[0].q[1] = *(const u32x4*)(kp0 + 8);                            \
        dst[1].q[0] = *(const u32x4*)(kp0 + 32);                           \
        dst[1].q[1] = *(const u32x4*)(kp0 + 40);                           \
        const u16* kp1 = kp0 + 16 * HD_;                                   \
        dst[2].q[0] = *(const u32x4*)(kp1);                                \
        dst[2].q[1] = *(const u32x4*)(kp1 + 8);                            \
        dst[3].q[0] = *(const u32x4*)(kp1 + 32);                           \
        dst[3].q[1] = *(const u32x4*)(kp1 + 40); } while (0)

    Frag16 kfa[4], kfb[4];
    LOAD_KF(kfa, 0);
    const int kend = (qb + 1) * 64;               // uniform across block (masked tail)
    for (int kb = 0; kb < kend; kb += 64) {
        LOAD_KF(kfb, kb + 32);
        attn_step(kb, qrow0, bh, wave, half, l16, qf0, qf1, kfa,
                  Vt, Ps[0], mrow, lrow, o);
        if (kb + 64 < kend) LOAD_KF(kfa, kb + 64);
        attn_step(kb + 32, qrow0, bh, wave, half, l16, qf0, qf1, kfb,
                  Vt, Ps[1], mrow, lrow, o);
    }
#undef LOAD_KF

    // normalize + store (bf16, token-major [M][C] with head offset)
#pragma unroll
    for (int r = 0; r < 8; ++r) {
        float inv = 1.0f / lrow[r];
        size_t row = (size_t)b * T_ + qrow0 + r + half * 8;
#pragma unroll
        for (int t = 0; t < 4; ++t)
            Y[row * C_ + h * HD_ + t * 16 + l16] = f2bf(o[t][r] * inv);
    }
}

// ---------------- final LN + 18-way head, wave per selected token ----------------
__global__ void __launch_bounds__(32) head_k(const float* __restrict__ x,
                                             const float* __restrict__ w,
                                             const float* __restrict__ bsh,
                                             const float* __restrict__ HW,
                                             float* __restrict__ out) {
    int lane = threadIdx.x;
    int gb = blockIdx.x;                 // 0..B*TS-1
    int b = gb / TS_, j = gb % TS_;
    const float* xr = x + ((size_t)b * T_ + 3 * j) * C_;
    float vals[24];
    float s = 0.0f;
#pragma unroll
    for (int i = 0; i < 24; ++i) { vals[i] = xr[lane + 32 * i]; s += vals[i]; }
#pragma unroll
    for (int d = 1; d < 32; d <<= 1) s += __shfl_xor(s, d, 32);
    float mean = s * (1.0f / (float)C_);
    float q = 0.0f;
#pragma unroll
    for (int i = 0; i < 24; ++i) { float dd = vals[i] - mean; q += dd * dd; }
#pragma unroll
    for (int d = 1; d < 32; d <<= 1) q += __shfl_xor(q, d, 32);
    float rstd = rsqrtf(q * (1.0f / (float)C_) + 1e-5f);
    float acc[NA_];
#pragma unroll
    for (int a = 0; a < NA_; ++a) acc[a] = 0.0f;
#pragma unroll
    for (int i = 0; i < 24; ++i) {
        int c = lane + 32 * i;
        float nx = (vals[i] - mean) * rstd * w[c] + bsh[c];
#pragma unroll
        for (int a = 0; a < NA_; ++a) acc[a] += nx * HW[a * C_ + c];
    }
#pragma unroll
    for (int a = 0; a < NA_; ++a) {
        float t = acc[a];
#pragma unroll
        for (int d = 1; d < 32; d <<= 1) t += __shfl_xor(t, d, 32);
        if (lane == 0) out[(size_t)gb * NA_ + a] = t;
    }
}

// ---------------- host orchestration ----------------
extern "C" void kernel_launch(void* const* d_in, const int* in_sizes, int n_in,
                              void* d_out, int out_size, void* d_ws, size_t ws_size,
                              hipStream_t stream) {
    const int*   states   = (const int*)  d_in[0];
    const int*   actions  = (const int*)  d_in[1];
    const int*   rewards  = (const int*)  d_in[2];
    const float* state_e  = (const float*)d_in[3];
    const float* reward_e = (const float*)d_in[4];
    const float* action_e = (const float*)d_in[5];
    const float* proj_w   = (const float*)d_in[6];
    const float* proj_b   = (const float*)d_in[7];
    const float* ln1_w    = (const float*)d_in[8];
    const float* ln1_b    = (const float*)d_in[9];
    const float* attn_w   = (const float*)d_in[10];
    const float* attn_b   = (const float*)d_in[11];
    const float* apr_w    = (const float*)d_in[12];
    const float* apr_b    = (const float*)d_in[13];
    const float* ln2_w    = (const float*)d_in[14];
    const float* ln2_b    = (const float*)d_in[15];
    const float* fc_w     = (const float*)d_in[16];
    const float* fc_b     = (const float*)d_in[17];
    const float* fp_w     = (const float*)d_in[18];
    const float* fp_b     = (const float*)d_in[19];
    const float* lnf_w    = (const float*)d_in[20];
    const float* lnf_b    = (const float*)d_in[21];
    const float* head_w   = (const float*)d_in[22];
    float* out = (float*)d_out;
    (void)in_sizes; (void)n_in; (void)out_size; (void)ws_size;

    size_t off = 0;
    auto take = [&](size_t bytes) -> char* {
        off = (off + 255) & ~(size_t)255;
        char* p = (char*)d_ws + off;
        off += bytes;
        return p;
    };

    u16* projW = (u16*)take((size_t)C_ * NT_ * 2);
    u16* attnW = (u16*)take((size_t)L_ * 3 * C_ * C_ * 2);
    u16* apW   = (u16*)take((size_t)L_ * C_ * C_ * 2);
    u16* fcW   = (u16*)take((size_t)L_ * 4 * C_ * C_ * 2);
    u16* fpW   = (u16*)take((size_t)L_ * C_ * 4 * C_ * 2);
    u16* embS  = (u16*)take((size_t)1 * NT_ * 2);
    u16* embA  = (u16*)take((size_t)NA_ * NT_ * 2);
    u16* embR  = (u16*)take((size_t)2 * NT_ * 2);
    u16* seqb  = (u16*)take((size_t)M_ * NT_ * 2);
    float* x   = (float*)take((size_t)M_ * C_ * 4);
    u16* hb    = (u16*)take((size_t)M_ * C_ * 2);
    u16* scr   = (u16*)take((size_t)M_ * 4 * C_ * 2);   // qkv (3C) / MLP hh (4C) alias
    u16* qkvb  = scr;
    u16* hh    = scr;
    u16* Qb    = (u16*)take((size_t)M_ * C_ * 2);       // [bh][T][HD]
    u16* Kb    = (u16*)take((size_t)M_ * C_ * 2);       // [bh][T][HD]
    u16* Vt    = (u16*)take((size_t)M_ * C_ * 2);       // [bh][HD][T] (transposed)
    u16* attno = (u16*)take((size_t)M_ * C_ * 2);

    auto conv = [&](const float* s, u16* d, size_t n) {
        convf2bf<<<dim3((unsigned)((n + 255) / 256)), dim3(256), 0, stream>>>(s, d, (int)n);
    };
    conv(proj_w,   projW, (size_t)C_ * NT_);
    conv(attn_w,   attnW, (size_t)L_ * 3 * C_ * C_);
    conv(apr_w,    apW,   (size_t)L_ * C_ * C_);
    conv(fc_w,     fcW,   (size_t)L_ * 4 * C_ * C_);
    conv(fp_w,     fpW,   (size_t)L_ * C_ * 4 * C_);
    conv(state_e,  embS,  (size_t)NT_);
    conv(action_e, embA,  (size_t)NA_ * NT_);
    conv(reward_e, embR,  (size_t)2 * NT_);

    embed_k<<<dim3(M_), dim3(128), 0, stream>>>(states, actions, rewards,
                                                embS, embA, embR, seqb);

    auto gemm = [&](const u16* A, const u16* W, const float* bias, const float* resid,
                    float* oF, u16* oH, int N, int K, int gel) {
        gemm_bf16<<<dim3(N / 128, M_ / 128), dim3(256), 0, stream>>>(
            A, W, bias, resid, oF, oH, N, K, gel);
    };

    // token embedding projection: x = seq @ proj_w^T + proj_b
    gemm(seqb, projW, proj_b, nullptr, x, nullptr, C_, NT_, 0);

    for (int l = 0; l < L_; ++l) {
        ln_k<<<dim3(M_ / 8), dim3(256), 0, stream>>>(x, ln1_w + l * C_, ln1_b + l * C_, hb);
        gemm(hb, attnW + (size_t)l * 3 * C_ * C_, attn_b + l * 3 * C_,
             nullptr, nullptr, qkvb, 3 * C_, C_, 0);
        rope_k<<<dim3(T_, H_, B_), dim3(64), 0, stream>>>(qkvb, Qb, Kb, Vt);
        attn_k<<<dim3(T_ / 64, H_, B_), dim3(128), 0, stream>>>(Qb, Kb, Vt, attno);
        gemm(attno, apW + (size_t)l * C_ * C_, apr_b + l * C_,
             x, x, nullptr, C_, C_, 0);
        ln_k<<<dim3(M_ / 8), dim3(256), 0, stream>>>(x, ln2_w + l * C_, ln2_b + l * C_, hb);
        gemm(hb, fcW + (size_t)l * 4 * C_ * C_, fc_b + l * 4 * C_,
             nullptr, nullptr, hh, 4 * C_, C_, 1);
        gemm(hh, fpW + (size_t)l * C_ * 4 * C_, fp_b + l * C_,
             x, x, nullptr, C_, 4 * C_, 0);
    }

    head_k<<<dim3(B_ * TS_), dim3(32), 0, stream>>>(x, lnf_w, lnf_b, head_w, out);
}
